// SmallTfE_9560597200957
// MI455X (gfx1250) — matrix operand order repaired and checked by reading, not compiled, not run on hardware
//
#include <hip/hip_runtime.h>

// Problem constants from the reference: N=8, L=2048, E=1024, V=512.
#define NB 8
#define LL 2048
#define EE 1024
#define VV 512
#define NCHUNK 16
#define CHUNK_M (LL / NCHUNK) // 128

typedef __attribute__((ext_vector_type(16))) _Float16 v16h;
typedef __attribute__((ext_vector_type(8)))  _Float16 v8h;
typedef __attribute__((ext_vector_type(8)))  float    v8f;

// ---------------------------------------------------------------------------
// Kernel 1: q_last[n,f] = sum_e x[n,L-1,e] * Wa_w[f,e] + Wa_b[f]
// x_last (8x1024 = 32KB) cached in LDS; each thread owns one f and all 8 n.
// Accumulate in f32; store as f16 (sole consumer is the f16 WMMA score pass).
// ---------------------------------------------------------------------------
__global__ __launch_bounds__(256) void k_qlast(const int* __restrict__ tokens,
                                               const float* __restrict__ Wa_w,
                                               const float* __restrict__ Wa_b,
                                               _Float16* __restrict__ Qh) {
  __shared__ float xs[NB * EE]; // 32 KB
  const int tid = threadIdx.x;
  for (int i = tid; i < NB * EE; i += 256) {
    const int n = i >> 10, e = i & 1023;
    xs[i] = (float)tokens[((size_t)n * LL + (LL - 1)) * EE + e];
  }
  __syncthreads();
  const int f = blockIdx.x * 256 + tid;
  float acc[NB];
#pragma unroll
  for (int n = 0; n < NB; ++n) acc[n] = 0.f;
  const float* wrow = Wa_w + (size_t)f * EE;
  for (int e = 0; e < EE; ++e) {
    const float w = wrow[e];
#pragma unroll
    for (int n = 0; n < NB; ++n) acc[n] += w * xs[n * EE + e];
  }
#pragma unroll
  for (int n = 0; n < NB; ++n) Qh[n * EE + f] = (_Float16)(acc[n] + Wa_b[f]);
}

// ---------------------------------------------------------------------------
// Kernel 2: scores[n,m] = q_last[n,:] . x[n,m,:]  via v_wmma_f32_16x16x32_f16.
// One wave per (batch, 16-row m-tile). Tokens are 0/1 -> exact in f16.
// A layout (16-bit A 16x32): lane<16 holds K 0-7,16-23; lane>=16 K 8-15,24-31.
// B: q chunk lives in column 0 (lanes 0 and 16); built branchlessly via select.
// D column 0: lane 0 = M 0-7, lane 16 = M 8-15.
// ---------------------------------------------------------------------------
__global__ __launch_bounds__(32) void k_scores(const int* __restrict__ tokens,
                                               const _Float16* __restrict__ Qh,
                                               float* __restrict__ S) {
  const int n     = blockIdx.y;
  const int mBase = blockIdx.x * 16;
  const int lane  = threadIdx.x;
  const int mrow  = lane & 15;
  const int hi    = lane >> 4;
  const bool col0 = (lane & 15) == 0;
  const int* __restrict__ xrow = tokens + ((size_t)n * LL + mBase + mrow) * EE;
  const _Float16* __restrict__ qh = Qh + n * EE;

  const v16h zero = {};
  v8f c = {};
  for (int k0 = 0; k0 < EE; k0 += 32) {
    const int ka = k0 + (hi ? 8 : 0);
    __builtin_prefetch(xrow + ka + 64, 0, 3); // global_prefetch_b8 (speculative)
    v16h a;
#pragma unroll
    for (int j = 0; j < 8; ++j) a[j]     = (_Float16)(float)xrow[ka + j];
#pragma unroll
    for (int j = 0; j < 8; ++j) a[8 + j] = (_Float16)(float)xrow[ka + 16 + j];

    const int kb = k0 + (hi ? 8 : 0);
    const v8h qlo = *(const v8h*)(qh + kb);      // K kb..kb+7   (16B aligned)
    const v8h qhi = *(const v8h*)(qh + kb + 16); // K kb+16..+23 (16B aligned)
    const v16h qv = __builtin_shufflevector(qlo, qhi, 0, 1, 2, 3, 4, 5, 6, 7,
                                            8, 9, 10, 11, 12, 13, 14, 15);
    const v16h b  = col0 ? qv : zero;        // branchless: keeps EXEC all-1s

    c = __builtin_amdgcn_wmma_f32_16x16x32_f16(false, a, false, b, (short)0, c,
                                               false, false);
  }
  if (lane == 0) {
#pragma unroll
    for (int r = 0; r < 8; ++r) S[n * LL + mBase + r] = c[r];
  } else if (lane == 16) {
#pragma unroll
    for (int r = 0; r < 8; ++r) S[n * LL + mBase + 8 + r] = c[r];
  }
}

// ---------------------------------------------------------------------------
// Kernel 3: in-place softmax over m (one block per batch, 2048 values).
// ---------------------------------------------------------------------------
__global__ __launch_bounds__(256) void k_softmax(float* __restrict__ S) {
  const int n   = blockIdx.x;
  const int tid = threadIdx.x;
  float* s = S + n * LL;
  __shared__ float red[256];

  float v[8];
  float lmax = -3.4e38f;
#pragma unroll
  for (int i = 0; i < 8; ++i) {
    v[i] = s[tid + i * 256];
    lmax = fmaxf(lmax, v[i]);
  }
  red[tid] = lmax;
  __syncthreads();
  for (int off = 128; off > 0; off >>= 1) {
    if (tid < off) red[tid] = fmaxf(red[tid], red[tid + off]);
    __syncthreads();
  }
  const float gmax = red[0];
  __syncthreads();

  float lsum = 0.f;
#pragma unroll
  for (int i = 0; i < 8; ++i) {
    v[i] = __expf(v[i] - gmax);
    lsum += v[i];
  }
  red[tid] = lsum;
  __syncthreads();
  for (int off = 128; off > 0; off >>= 1) {
    if (tid < off) red[tid] += red[tid + off];
    __syncthreads();
  }
  const float inv = 1.f / red[0];
#pragma unroll
  for (int i = 0; i < 8; ++i) s[tid + i * 256] = v[i] * inv;
}

// ---------------------------------------------------------------------------
// Kernel 4: a[n,e] = sum_m w[n,m] * x[n,m,e] — coalesced int4 stream.
// Deterministic per-chunk partials (no float atomics), reduced by kernel 4b.
// Second pass over tokens; whole tensor (64 MB) fits in the 192 MB L2.
// ---------------------------------------------------------------------------
__global__ __launch_bounds__(256) void k_attn_partial(const int* __restrict__ tokens,
                                                      const float* __restrict__ W,
                                                      float* __restrict__ P) {
  const int chunk = blockIdx.x;
  const int n     = blockIdx.y;
  const int tid   = threadIdx.x;
  __shared__ float wl[CHUNK_M];
  if (tid < CHUNK_M) wl[tid] = W[n * LL + chunk * CHUNK_M + tid];
  __syncthreads();

  const int e0 = tid * 4;
  float4 acc = {0.f, 0.f, 0.f, 0.f};
  const int* __restrict__ base =
      tokens + ((size_t)n * LL + (size_t)chunk * CHUNK_M) * EE + e0;
  for (int m = 0; m < CHUNK_M; ++m) {
    const int4 xv = *(const int4*)(base + (size_t)m * EE);
    const float wm = wl[m];
    acc.x += wm * (float)xv.x;
    acc.y += wm * (float)xv.y;
    acc.z += wm * (float)xv.z;
    acc.w += wm * (float)xv.w;
  }
  *(float4*)(P + ((size_t)(chunk * NB + n)) * EE + e0) = acc;
}

// Reduce partials into Ah (f16: sole consumer is the f16 WMMA logits GEMM).
__global__ __launch_bounds__(256) void k_attn_reduce(const float* __restrict__ P,
                                                     _Float16* __restrict__ Ah) {
  const int idx = blockIdx.x * 256 + threadIdx.x; // 0..8191
  const int n = idx >> 10, e = idx & 1023;
  float s = 0.f;
#pragma unroll
  for (int c = 0; c < NCHUNK; ++c) s += P[((size_t)(c * NB + n)) * EE + e];
  Ah[idx] = (_Float16)s;
}

// ---------------------------------------------------------------------------
// Kernel 5: logits[n,v] = sum_e a[n,e] * Wb_w[v,e] + Wb_b[v].
// Real GEMM: M=8 (zero-padded to 16) x N=512 x K=1024, WMMA f16, f32 acc.
// One wave per 16-wide v-tile; D rows 0..7 (lanes 0..15) are the 8 batches.
// Row padding done branchlessly (clamped address + value select).
// ---------------------------------------------------------------------------
__global__ __launch_bounds__(32) void k_logits(const _Float16* __restrict__ Ah,
                                               const float* __restrict__ Wb_w,
                                               const float* __restrict__ Wb_b,
                                               float* __restrict__ out) {
  const int vBase = blockIdx.x * 16;
  const int lane  = threadIdx.x;
  const int hi    = lane >> 4;
  const int mrow  = lane & 15; // batch row in A-tile (valid if < NB)
  const int col   = lane & 15; // output column in B/D-tile
  const bool rowOK = mrow < NB;

  const _Float16* __restrict__ arow = Ah + (size_t)(rowOK ? mrow : 0) * EE;
  const float*    __restrict__ brow = Wb_w + (size_t)(vBase + col) * EE;

  const v16h zero = {};
  v8f c = {};
  for (int k0 = 0; k0 < EE; k0 += 32) {
    const int ka = k0 + (hi ? 8 : 0);
    const v8h lo  = *(const v8h*)(arow + ka);      // K ka..ka+7   (16B aligned)
    const v8h hi8 = *(const v8h*)(arow + ka + 16); // K ka+16..+23 (16B aligned)
    const v16h av = __builtin_shufflevector(lo, hi8, 0, 1, 2, 3, 4, 5, 6, 7,
                                            8, 9, 10, 11, 12, 13, 14, 15);
    const v16h a = rowOK ? av : zero; // branchless zero-padding of rows 8..15

    const int kb = k0 + (hi ? 8 : 0);
    v16h b;
#pragma unroll
    for (int j = 0; j < 16; ++j) b[j] = (_Float16)brow[kb + j + (j >= 8 ? 8 : 0)];

    c = __builtin_amdgcn_wmma_f32_16x16x32_f16(false, a, false, b, (short)0, c,
                                               false, false);
  }
  if (lane < 16) {
#pragma unroll
    for (int r = 0; r < 8; ++r)
      out[(size_t)r * VV + vBase + lane] = c[r] + Wb_b[vBase + lane];
  }
}

// ---------------------------------------------------------------------------
extern "C" void kernel_launch(void* const* d_in, const int* in_sizes, int n_in,
                              void* d_out, int out_size, void* d_ws, size_t ws_size,
                              hipStream_t stream) {
  (void)in_sizes; (void)n_in; (void)out_size; (void)ws_size;
  const int*   tokens = (const int*)d_in[0];
  const float* Wa_w   = (const float*)d_in[1];
  const float* Wa_b   = (const float*)d_in[2];
  const float* Wb_w   = (const float*)d_in[3];
  const float* Wb_b   = (const float*)d_in[4];
  float* out = (float*)d_out;

  char* ws = (char*)d_ws;
  _Float16* Qh = (_Float16*)(ws);             // 8*1024 f16 (16 KB)
  float*    S  = (float*)(ws + 32 * 1024);    // 8*2048 f32 (64 KB)
  _Float16* Ah = (_Float16*)(ws + 96 * 1024); // 8*1024 f16 (16 KB)
  float*    P  = (float*)(ws + 128 * 1024);   // 16*8*1024 f32 (512 KB)

  hipLaunchKernelGGL(k_qlast, dim3(EE / 256), dim3(256), 0, stream,
                     tokens, Wa_w, Wa_b, Qh);
  hipLaunchKernelGGL(k_scores, dim3(LL / 16, NB), dim3(32), 0, stream,
                     tokens, Qh, S);
  hipLaunchKernelGGL(k_softmax, dim3(NB), dim3(256), 0, stream, S);
  hipLaunchKernelGGL(k_attn_partial, dim3(NCHUNK, NB), dim3(256), 0, stream,
                     tokens, S, P);
  hipLaunchKernelGGL(k_attn_reduce, dim3(NB * EE / 256), dim3(256), 0, stream,
                     P, Ah);
  hipLaunchKernelGGL(k_logits, dim3(VV / 16), dim3(32), 0, stream,
                     Ah, Wb_w, Wb_b, out);
}
